// BatchNeuralMemoryV2_47287589929766
// MI455X (gfx1250) — matrix operand order repaired and checked by reading, not compiled, not run on hardware
//
#include <hip/hip_runtime.h>
#include <hip/hip_bf16.h>

// ---------------- problem constants ----------------
constexpr int DD    = 1024;
constexpr int INTR  = 2048;
constexpr int BB    = 8;
constexpr int SS    = 4096;
constexpr int CC    = 64;
constexpr int NCH   = 64;          // SS / CC
constexpr int RR    = 512;         // BB * CC rows per chunk
constexpr float EPSF = 1e-6f;

typedef __attribute__((ext_vector_type(16))) __bf16 v16bf;
typedef __attribute__((ext_vector_type(8)))  float  v8f;

union FragBF { v16bf v; uint4 q[2]; };

// ---------------- small device helpers ----------------
__device__ __forceinline__ unsigned short f32_to_bf16(float f) {
  unsigned int u = __builtin_bit_cast(unsigned int, f);
  unsigned int r = (u + 0x7FFFu + ((u >> 16) & 1u)) >> 16;
  return (unsigned short)r;
}
__device__ __forceinline__ float bf16_to_f32(unsigned short h) {
  unsigned int u = ((unsigned int)h) << 16;
  return __builtin_bit_cast(float, u);
}
__device__ __forceinline__ float geluf(float x) {
  return 0.5f * x * (1.0f + erff(x * 0.70710678118654752f));
}
__device__ __forceinline__ float sigmoidf(float x) { return 1.0f / (1.0f + expf(-x)); }

__device__ float block_reduce_sum(float v) {
  __shared__ float red[256];
  red[threadIdx.x] = v;
  __syncthreads();
  for (int s = 128; s > 0; s >>= 1) {
    if ((int)threadIdx.x < s) red[threadIdx.x] += red[threadIdx.x + s];
    __syncthreads();
  }
  float r = red[0];
  __syncthreads();
  return r;
}

// ---------------- WMMA GEMM: C[M,N] = A[M,K] * B[N,K]^T (bf16 in, f32 out) ----------------
// Register-tiled: each wave owns a 64(M) x 32(N) output tile = 4x2 grid of 16x16
// WMMA accumulators (64 VGPRs). K is a template constant so all row offsets
// (i*16*K*2 bytes, max 192 KB < 24-bit IOFFSET) fold into instruction immediates:
// one base pointer per operand, no pointer spills, ~130 live VGPRs total.
// Per 32-deep K step: 6 KB of L2 traffic feeds 8 WMMAs = 21 FLOP/byte.
// Block = 256 thr = 8 waves (2x4) = 128(M) x 128(N).
// Requires M % 128 == 0, N % 128 == 0, K % 64 == 0.
template <int Kc>
__global__ void gemm_bf16_nt(const unsigned short* __restrict__ A,
                             const unsigned short* __restrict__ B,
                             float* __restrict__ C, int M, int N) {
  const int lane = threadIdx.x & 31;
  const int wave = threadIdx.x >> 5;       // 0..7
  const int wm = wave >> 2, wn = wave & 3; // 2 x 4 wave grid
  const int mBase = blockIdx.y * 128 + wm * 64;
  const int nBase = blockIdx.x * 128 + wn * 32;
  const int half = lane >> 4;              // lane half selects K runs (ISA 7.12.2)
  const int l15  = lane & 15;

  // A: K runs {0..7,16..23} / {8..15,24..31}; B: K runs {0..7,16..23} / {8..15,24..31}
  const unsigned short* abase = A + (size_t)(mBase + l15) * Kc + half * 8;
  const unsigned short* bbase = B + (size_t)(nBase + l15) * Kc + half * 8;

  v8f acc[4][2] = {};

#pragma unroll 2
  for (int k = 0; k < Kc; k += 32) {
    FragBF a[4], b[2];
#pragma unroll
    for (int i = 0; i < 4; ++i) {
      a[i].q[0] = *(const uint4*)(abase + i * 16 * Kc + k);
      a[i].q[1] = *(const uint4*)(abase + i * 16 * Kc + k + 16);
    }
#pragma unroll
    for (int j = 0; j < 2; ++j) {
      b[j].q[0] = *(const uint4*)(bbase + j * 16 * Kc + k);
      b[j].q[1] = *(const uint4*)(bbase + j * 16 * Kc + k + 16);
    }
    if (k + 32 < Kc) {                      // gfx1250 global_prefetch_b8
#pragma unroll
      for (int i = 0; i < 4; ++i) __builtin_prefetch(abase + i * 16 * Kc + k + 32, 0, 1);
#pragma unroll
      for (int j = 0; j < 2; ++j) __builtin_prefetch(bbase + j * 16 * Kc + k + 32, 0, 1);
    }
#pragma unroll
    for (int i = 0; i < 4; ++i)
#pragma unroll
      for (int j = 0; j < 2; ++j)
        acc[i][j] = __builtin_amdgcn_wmma_f32_16x16x32_bf16(
            false, a[i].v, false, b[j].v, (short)0, acc[i][j], false, false);
  }

  // C/D layout: VGPR r -> row (tile + half*8 + r), col (tile + l15)
#pragma unroll
  for (int i = 0; i < 4; ++i) {
#pragma unroll
    for (int j = 0; j < 2; ++j) {
      float* crow = C + (size_t)(mBase + i * 16 + half * 8) * N + nBase + j * 16 + l15;
#pragma unroll
      for (int r = 0; r < 8; ++r) crow[(size_t)r * N] = acc[i][j][r];
    }
  }
}

// ---------------- elementwise / transpose / reductions ----------------
__global__ void cvt_bf16(unsigned short* __restrict__ dst, const float* __restrict__ src, size_t n) {
  size_t i = (size_t)blockIdx.x * 256 + threadIdx.x;
  if (i < n) dst[i] = f32_to_bf16(src[i]);
}

__global__ void zerof(float* __restrict__ x, size_t n) {
  size_t i = (size_t)blockIdx.x * 256 + threadIdx.x;
  if (i < n) x[i] = 0.0f;
}

// dst[c*R + r] = bf16(src[r*C + c])  (LDS-tiled)
__global__ void transpose_f32_bf16(unsigned short* __restrict__ dst,
                                   const float* __restrict__ src, int R, int Cc) {
  __shared__ float tile[32][33];
  const int bx = blockIdx.x * 32, by = blockIdx.y * 32;
  const int tx = threadIdx.x & 31, ty = threadIdx.x >> 5; // 256 = 32x8
  for (int i = ty; i < 32; i += 8) {
    int r = by + i, c = bx + tx;
    tile[i][tx] = (r < R && c < Cc) ? src[(size_t)r * Cc + c] : 0.0f;
  }
  __syncthreads();
  for (int i = ty; i < 32; i += 8) {
    int c = bx + i, r = by + tx;
    if (c < Cc && r < R) dst[(size_t)c * R + r] = f32_to_bf16(tile[tx][i]);
  }
}

__global__ void gelu_fb(float* __restrict__ outf, unsigned short* __restrict__ outb,
                        const float* __restrict__ in, size_t n) {
  size_t i = (size_t)blockIdx.x * 256 + threadIdx.x;
  if (i >= n) return;
  float g = geluf(in[i]);
  if (outf) outf[i] = g;
  if (outb) outb[i] = f32_to_bf16(g);
}

// v = gelu(xw) into chunk-major layout [t][b*CC+c][d]
__global__ void gelu_chunk_f32(float* __restrict__ dst, const float* __restrict__ xw) {
  size_t i = (size_t)blockIdx.x * 256 + threadIdx.x;
  if (i >= (size_t)BB * SS * DD) return;
  int d = (int)(i % DD);
  size_t s = i / DD;
  int b = (int)(s >> 12), rem = (int)(s & 4095), t = rem >> 6, c = rem & 63;
  size_t o = ((size_t)t * RR + b * CC + c) * DD + d;
  dst[o] = geluf(xw[i]);
}

// k = rmsnorm(gelu(xw), knw) -> kbf chunk-major, kT chunk-transposed
__global__ void gelu_rms_k(const float* __restrict__ xw, const float* __restrict__ nw,
                           unsigned short* __restrict__ kbf, unsigned short* __restrict__ kT) {
  int s = blockIdx.x;
  int b = s >> 12, rem = s & 4095, t = rem >> 6, c = rem & 63;
  int rr = b * CC + c;
  const float* row = xw + (size_t)s * DD;
  float g[4], ss = 0.0f;
#pragma unroll
  for (int i = 0; i < 4; ++i) {
    int d = threadIdx.x + i * 256;
    g[i] = geluf(row[d]);
    ss += g[i] * g[i];
  }
  float inv = rsqrtf(block_reduce_sum(ss) / (float)DD + EPSF);
#pragma unroll
  for (int i = 0; i < 4; ++i) {
    int d = threadIdx.x + i * 256;
    unsigned short h = f32_to_bf16(g[i] * inv * nw[d]);
    kbf[((size_t)t * RR + rr) * DD + d] = h;
    kT[(size_t)t * DD * RR + (size_t)d * RR + rr] = h;
  }
}

// q = rmsnorm(gelu(xw), qnw) -> qf (f32, row-major) and qbf
__global__ void gelu_rms_q(const float* __restrict__ xw, const float* __restrict__ nw,
                           float* __restrict__ qf, unsigned short* __restrict__ qbf) {
  int s = blockIdx.x;
  const float* row = xw + (size_t)s * DD;
  float g[4], ss = 0.0f;
#pragma unroll
  for (int i = 0; i < 4; ++i) {
    int d = threadIdx.x + i * 256;
    g[i] = geluf(row[d]);
    ss += g[i] * g[i];
  }
  float inv = rsqrtf(block_reduce_sum(ss) / (float)DD + EPSF);
#pragma unroll
  for (int i = 0; i < 4; ++i) {
    int d = threadIdx.x + i * 256;
    float v = g[i] * inv * nw[d];
    qf[(size_t)s * DD + d] = v;
    qbf[(size_t)s * DD + d] = f32_to_bf16(v);
  }
}

// gate dot products over C*D = 65536 elems: dots[w][b][t], w in {alpha,theta,eta}
__global__ void dots_kernel(const float* __restrict__ x, const float* __restrict__ aw,
                            const float* __restrict__ tw, const float* __restrict__ ew,
                            float* __restrict__ dots) {
  int t = blockIdx.x, b = blockIdx.y;
  const float* xp = x + (size_t)b * SS * DD + (size_t)t * CC * DD;
  float sa = 0, st = 0, se = 0;
  for (int i = threadIdx.x; i < CC * DD; i += 256) {
    float xv = xp[i];
    sa += xv * aw[i]; st += xv * tw[i]; se += xv * ew[i];
  }
  sa = block_reduce_sum(sa);
  st = block_reduce_sum(st);
  se = block_reduce_sum(se);
  if (threadIdx.x == 0) {
    dots[0 * BB * NCH + b * NCH + t] = sa;
    dots[1 * BB * NCH + b * NCH + t] = st;
    dots[2 * BB * NCH + b * NCH + t] = se;
  }
}

__global__ void finish_gates(const float* __restrict__ dots, float* __restrict__ alpha,
                             float* __restrict__ eta, float* __restrict__ theta) {
  int t = threadIdx.x;
  if (t >= NCH) return;
  float sa = 0, se = 0;
  for (int b = 0; b < BB; ++b) {
    sa += dots[0 * BB * NCH + b * NCH + t];
    se += dots[2 * BB * NCH + b * NCH + t];
    theta[t * BB + b] = sigmoidf(dots[1 * BB * NCH + b * NCH + t]) * 0.01f;
  }
  alpha[t] = sigmoidf(sa / (float)BB);
  eta[t]   = sigmoidf(se / (float)BB);
}

// rms backward for one chunk row: produces dh3 (+bf16) and accumulates dgamma
__global__ void rms_backward(const float* __restrict__ h3, const unsigned short* __restrict__ kbf_t,
                             const float* __restrict__ v_t, const float* __restrict__ gamma,
                             const float* __restrict__ theta_t, float* __restrict__ dgamma,
                             float* __restrict__ dh3f, unsigned short* __restrict__ dh3b) {
  int r = blockIdx.x;
  int b = r >> 6;                 // r = b*CC + c
  float th = theta_t[b];
  const float* h3r = h3 + (size_t)r * DD;
  const unsigned short* kr = kbf_t + (size_t)r * DD;
  const float* vr = v_t + (size_t)r * DD;
  float h3v[4], ss = 0.0f;
#pragma unroll
  for (int i = 0; i < 4; ++i) {
    int d = threadIdx.x + i * 256;
    h3v[i] = h3r[d];
    ss += h3v[i] * h3v[i];
  }
  float inv = rsqrtf(block_reduce_sum(ss) / (float)DD + EPSF);
  const float coefN = -2.0f / (float)(RR * DD);
  float h3n[4], dh3n[4], s2 = 0.0f;
#pragma unroll
  for (int i = 0; i < 4; ++i) {
    int d = threadIdx.x + i * 256;
    h3n[i] = h3v[i] * inv;
    float g = gamma[d];
    float pred = bf16_to_f32(kr[d]) + h3n[i] * g;
    float dl = coefN * (vr[d] - pred) * th;
    atomicAdd(&dgamma[d], dl * h3n[i]);
    dh3n[i] = dl * g;
    s2 += dh3n[i] * h3n[i];
  }
  float coeff = block_reduce_sum(s2) / (float)DD;
#pragma unroll
  for (int i = 0; i < 4; ++i) {
    int d = threadIdx.x + i * 256;
    float dv = inv * (dh3n[i] - h3n[i] * coeff);
    dh3f[(size_t)r * DD + d] = dv;
    dh3b[(size_t)r * DD + d] = f32_to_bf16(dv);
  }
}

// dh1 = dh2 * gelu'(h1)
__global__ void gelu_bwd(float* __restrict__ dh1, const float* __restrict__ dh2,
                         const float* __restrict__ h1, size_t n) {
  size_t i = (size_t)blockIdx.x * 256 + threadIdx.x;
  if (i >= n) return;
  float h = h1[i];
  float cdf = 0.5f * (1.0f + erff(h * 0.70710678118654752f));
  float pdf = 0.3989422804014327f * expf(-0.5f * h * h);
  dh1[i] = dh2[i] * (cdf + h * pdf);
}

__global__ void sqsum(const float* __restrict__ x, size_t n, float* __restrict__ accum) {
  float s = 0.0f;
  for (size_t i = (size_t)blockIdx.x * 256 + threadIdx.x; i < n; i += (size_t)gridDim.x * 256) {
    float v = x[i];
    s += v * v;
  }
  s = block_reduce_sum(s);
  if (threadIdx.x == 0) atomicAdd(accum, s);
}

// momentum + decay update with global-norm clip; refreshes bf16 (and transposed) copies
__global__ void update_w(float* __restrict__ W, float* __restrict__ sW,
                         const float* __restrict__ dW, unsigned short* __restrict__ Wbf,
                         unsigned short* __restrict__ WT, const float* __restrict__ normsq,
                         const float* __restrict__ alpha, const float* __restrict__ eta,
                         int t, int rows, int cols) {
  size_t n = (size_t)rows * cols;
  size_t i = (size_t)blockIdx.x * 256 + threadIdx.x;
  if (i >= n) return;
  float total = sqrtf(*normsq);
  float clip = fminf(1.0f, 1.0f / (total + 1e-6f));
  float beta = 1.0f - alpha[t];
  float et = eta[t];
  float s = et * sW[i] - dW[i] * clip;
  float w = beta * W[i] + s;
  sW[i] = s;
  W[i] = w;
  if (Wbf) Wbf[i] = f32_to_bf16(w);
  if (WT) {
    int rr = (int)(i / cols), cc = (int)(i % cols);
    WT[(size_t)cc * rows + rr] = f32_to_bf16(w);
  }
}

__global__ void init_w(float* __restrict__ W, float* __restrict__ sW,
                       unsigned short* __restrict__ Wbf, unsigned short* __restrict__ WT,
                       const float* __restrict__ src, int rows, int cols) {
  size_t n = (size_t)rows * cols;
  size_t i = (size_t)blockIdx.x * 256 + threadIdx.x;
  if (i >= n) return;
  float w = src[i];
  W[i] = w;
  sW[i] = 0.0f;
  if (Wbf) Wbf[i] = f32_to_bf16(w);
  if (WT) {
    int rr = (int)(i / cols), cc = (int)(i % cols);
    WT[(size_t)cc * rows + rr] = f32_to_bf16(w);
  }
}

// out = q + rmsnorm(h3) * gamma  (per row; pointers pre-offset to chunk)
__global__ void final_out_kernel(float* __restrict__ out, const float* __restrict__ q,
                                 const float* __restrict__ h3, const float* __restrict__ gamma) {
  int r = blockIdx.x;
  const float* h3r = h3 + (size_t)r * DD;
  float h3v[4], ss = 0.0f;
#pragma unroll
  for (int i = 0; i < 4; ++i) {
    int d = threadIdx.x + i * 256;
    h3v[i] = h3r[d];
    ss += h3v[i] * h3v[i];
  }
  float inv = rsqrtf(block_reduce_sum(ss) / (float)DD + EPSF);
#pragma unroll
  for (int i = 0; i < 4; ++i) {
    int d = threadIdx.x + i * 256;
    size_t o = (size_t)r * DD + d;
    out[o] = q[o] + h3v[i] * inv * gamma[d];
  }
}

// ---------------- host orchestration ----------------
extern "C" void kernel_launch(void* const* d_in, const int* in_sizes, int n_in,
                              void* d_out, int out_size, void* d_ws, size_t ws_size,
                              hipStream_t stream) {
  const float* x    = (const float*)d_in[0];
  const float* wq   = (const float*)d_in[1];
  const float* wk   = (const float*)d_in[2];
  const float* wv   = (const float*)d_in[3];
  const float* qnw  = (const float*)d_in[4];
  const float* knw  = (const float*)d_in[5];
  const float* aw   = (const float*)d_in[6];
  const float* tw   = (const float*)d_in[7];
  const float* ew   = (const float*)d_in[8];
  const float* W0in = (const float*)d_in[9];
  const float* W1in = (const float*)d_in[10];
  const float* lnw  = (const float*)d_in[11];
  float* out = (float*)d_out;

  char* p = (char*)d_ws;
  auto alloc = [&](size_t bytes) -> void* {
    char* r = p;
    p += (bytes + 255) & ~(size_t)255;
    return (void*)r;
  };
  const size_t NXD = (size_t)BB * SS * DD;

  unsigned short* xbf  = (unsigned short*)alloc(NXD * 2);
  unsigned short* wqb  = (unsigned short*)alloc((size_t)DD * DD * 2);
  unsigned short* wkb  = (unsigned short*)alloc((size_t)DD * DD * 2);
  unsigned short* wvb  = (unsigned short*)alloc((size_t)DD * DD * 2);
  float*          xw   = (float*)alloc(NXD * 4);
  unsigned short* kbf  = (unsigned short*)alloc(NXD * 2);
  unsigned short* kT   = (unsigned short*)alloc(NXD * 2);
  float*          vch  = (float*)alloc(NXD * 4);
  float*          qf   = (float*)alloc(NXD * 4);
  unsigned short* qbf  = (unsigned short*)alloc(NXD * 2);

  const size_t NW = (size_t)INTR * DD;
  float* W0f = (float*)alloc(NW * 4);
  float* W1f = (float*)alloc(NW * 4);
  float* sW0 = (float*)alloc(NW * 4);
  float* sW1 = (float*)alloc(NW * 4);
  unsigned short* W0bf = (unsigned short*)alloc(NW * 2);
  unsigned short* W1bf = (unsigned short*)alloc(NW * 2);
  unsigned short* W1T  = (unsigned short*)alloc(NW * 2);
  float* gammaf = (float*)alloc(DD * 4);
  float* sg     = (float*)alloc(DD * 4);

  float* dots   = (float*)alloc(3 * BB * NCH * 4);
  float* alphaD = (float*)alloc(NCH * 4);
  float* etaD   = (float*)alloc(NCH * 4);
  float* thetaD = (float*)alloc(NCH * BB * 4);

  const size_t NRI = (size_t)RR * INTR;   // 512*2048
  const size_t NRD = (size_t)RR * DD;     // 512*1024
  float* h1   = (float*)alloc(NRI * 4);
  float* h2f  = (float*)alloc(NRI * 4);
  unsigned short* h2b = (unsigned short*)alloc(NRI * 2);
  unsigned short* h2T = (unsigned short*)alloc(NRI * 2);
  float* h3   = (float*)alloc(NRD * 4);
  float* dh3  = (float*)alloc(NRD * 4);
  unsigned short* dh3b = (unsigned short*)alloc(NRD * 2);
  unsigned short* dh3T = (unsigned short*)alloc(NRD * 2);
  float* dh2  = (float*)alloc(NRI * 4);
  float* dh1  = (float*)alloc(NRI * 4);
  unsigned short* dh1T = (unsigned short*)alloc(NRI * 2);
  float* dW0  = (float*)alloc(NW * 4);
  float* dW1  = (float*)alloc(NW * 4);
  float* dgamma = (float*)alloc((DD + 1) * 4);   // + normsq scalar (contiguous)
  float* normsq = dgamma + DD;

  const int CHM = 4096;                    // query-pass row chunk
  float* hC  = (float*)alloc((size_t)CHM * INTR * 4);
  unsigned short* hCb = (unsigned short*)alloc((size_t)CHM * INTR * 2);
  float* h3C = (float*)alloc((size_t)CHM * DD * 4);

  auto blocks = [](size_t n) { return (unsigned)((n + 255) / 256); };
  auto gemmNT = [&](const unsigned short* A, const unsigned short* B, float* Cc,
                    int M, int N, int K) {
    dim3 g(N / 128, M / 128);
    if (K == 512)
      gemm_bf16_nt<512><<<g, 256, 0, stream>>>(A, B, Cc, M, N);
    else if (K == 1024)
      gemm_bf16_nt<1024><<<g, 256, 0, stream>>>(A, B, Cc, M, N);
    else
      gemm_bf16_nt<2048><<<g, 256, 0, stream>>>(A, B, Cc, M, N);
  };
  auto transp = [&](unsigned short* dst, const float* src, int R, int Cc) {
    dim3 g((Cc + 31) / 32, (R + 31) / 32);
    transpose_f32_bf16<<<g, 256, 0, stream>>>(dst, src, R, Cc);
  };

  // -------- stage A: conversions, gates, projections --------
  cvt_bf16<<<blocks(NXD), 256, 0, stream>>>(xbf, x, NXD);
  cvt_bf16<<<blocks(NW / 2), 256, 0, stream>>>(wqb, wq, (size_t)DD * DD);
  cvt_bf16<<<blocks(NW / 2), 256, 0, stream>>>(wkb, wk, (size_t)DD * DD);
  cvt_bf16<<<blocks(NW / 2), 256, 0, stream>>>(wvb, wv, (size_t)DD * DD);

  dots_kernel<<<dim3(NCH, BB), 256, 0, stream>>>(x, aw, tw, ew, dots);
  finish_gates<<<1, 64, 0, stream>>>(dots, alphaD, etaD, thetaD);

  gemmNT(xbf, wkb, xw, BB * SS, DD, DD);
  gelu_rms_k<<<BB * SS, 256, 0, stream>>>(xw, knw, kbf, kT);

  gemmNT(xbf, wvb, xw, BB * SS, DD, DD);
  gelu_chunk_f32<<<blocks(NXD), 256, 0, stream>>>(vch, xw);

  gemmNT(xbf, wqb, xw, BB * SS, DD, DD);
  gelu_rms_q<<<BB * SS, 256, 0, stream>>>(xw, qnw, qf, qbf);

  init_w<<<blocks(NW), 256, 0, stream>>>(W0f, sW0, W0bf, nullptr, W0in, INTR, DD);
  init_w<<<blocks(NW), 256, 0, stream>>>(W1f, sW1, W1bf, W1T, W1in, DD, INTR);
  init_w<<<blocks(DD), 256, 0, stream>>>(gammaf, sg, nullptr, nullptr, lnw, 1, DD);

  // -------- stage B: sequential memory scan (64 steps) --------
  for (int t = 0; t < NCH; ++t) {
    const unsigned short* kbf_t = kbf + (size_t)t * RR * DD;
    const unsigned short* kT_t  = kT + (size_t)t * DD * RR;
    const float* v_t = vch + (size_t)t * RR * DD;

    gemmNT(kbf_t, W0bf, h1, RR, INTR, DD);                       // h1 = k W0^T
    gelu_fb<<<blocks(NRI), 256, 0, stream>>>(h2f, h2b, h1, NRI); // h2 = gelu(h1)
    transp(h2T, h2f, RR, INTR);

    gemmNT(h2b, W1bf, h3, RR, DD, INTR);                         // h3 = h2 W1^T

    zerof<<<blocks(DD + 1), 256, 0, stream>>>(dgamma, DD + 1);
    rms_backward<<<RR, 256, 0, stream>>>(h3, kbf_t, v_t, gammaf, thetaD + t * BB,
                                         dgamma, dh3, dh3b);

    transp(dh3T, dh3, RR, DD);
    gemmNT(dh3T, h2T, dW1, DD, INTR, RR);                        // dW1 = dh3^T h2
    gemmNT(dh3b, W1T, dh2, RR, INTR, DD);                        // dh2 = dh3 W1
    gelu_bwd<<<blocks(NRI), 256, 0, stream>>>(dh1, dh2, h1, NRI);
    transp(dh1T, dh1, RR, INTR);
    gemmNT(dh1T, kT_t, dW0, INTR, DD, RR);                       // dW0 = dh1^T k

    sqsum<<<512, 256, 0, stream>>>(dW0, NW, normsq);
    sqsum<<<512, 256, 0, stream>>>(dW1, NW, normsq);
    sqsum<<<4, 256, 0, stream>>>(dgamma, (size_t)DD, normsq);

    update_w<<<blocks(NW), 256, 0, stream>>>(W0f, sW0, dW0, W0bf, nullptr,
                                             normsq, alphaD, etaD, t, INTR, DD);
    update_w<<<blocks(NW), 256, 0, stream>>>(W1f, sW1, dW1, W1bf, W1T,
                                             normsq, alphaD, etaD, t, DD, INTR);
    update_w<<<blocks(DD), 256, 0, stream>>>(gammaf, sg, dgamma, nullptr, nullptr,
                                             normsq, alphaD, etaD, t, 1, DD);
  }

  // -------- stage C: query pass against final weights --------
  for (int ch = 0; ch < (BB * SS) / CHM; ++ch) {
    const size_t off = (size_t)ch * CHM * DD;
    gemmNT(qbf + off, W0bf, hC, CHM, INTR, DD);                  // h = q W0f^T
    gelu_fb<<<blocks((size_t)CHM * INTR), 256, 0, stream>>>(nullptr, hCb, hC,
                                                            (size_t)CHM * INTR);
    gemmNT(hCb, W1bf, h3C, CHM, DD, INTR);                       // h3 = h W1f^T
    final_out_kernel<<<CHM, 256, 0, stream>>>(out + off, qf + off, h3C, gammaf);
  }
  (void)in_sizes; (void)n_in; (void)out_size; (void)ws_size;
}
